// Encoder_59038620451500
// MI455X (gfx1250) — matrix operand order repaired and checked by reading, not compiled, not run on hardware
//
#include <hip/hip_runtime.h>
#include <hip/hip_bf16.h>

// ---------------- problem constants ----------------
#define NL    4
#define DM    768
#define NH    12
#define DKH   64
#define DFF   3072
#define NB    2
#define SEQ   2048
#define NTOK  (NB*SEQ)

typedef __attribute__((ext_vector_type(16))) _Float16 v16h;
typedef __attribute__((ext_vector_type(8)))  _Float16 v8h;
typedef __attribute__((ext_vector_type(8)))  float    v8f;

#define CAT16(lo,hi) __builtin_shufflevector((lo),(hi),0,1,2,3,4,5,6,7,8,9,10,11,12,13,14,15)
#define WMMA(a,b,c)  __builtin_amdgcn_wmma_f32_16x16x32_f16(false,(a),false,(b),(short)0,(c),false,false)

// ---- CDNA5 async global->LDS copy (ASYNCcnt-tracked, no VGPR staging) -------
__device__ __forceinline__ void async_ld_b128(void* lds_dst, const void* gsrc) {
    unsigned loff = (unsigned)(uintptr_t)lds_dst;   // low 32 bits of generic LDS addr == LDS offset
    asm volatile("global_load_async_to_lds_b128 %0, %1, off"
                 :: "v"(loff), "v"(gsrc)
                 : "memory");
}
__device__ __forceinline__ void wait_async0() {
#if __has_builtin(__builtin_amdgcn_s_wait_asynccnt)
    __builtin_amdgcn_s_wait_asynccnt(0);
#else
    asm volatile("s_wait_asynccnt 0x0" ::: "memory");
#endif
}

// ---------------- weight transpose+convert: src[K][M] f32 -> dst[M][K] f16 ----
__global__ void k_tr_f32h(const float* __restrict__ src, _Float16* __restrict__ dst,
                          int K, int M) {
    __shared__ _Float16 t[32][33];
    int tx = threadIdx.x, ty = threadIdx.y;
    int m0 = blockIdx.x * 32, k0 = blockIdx.y * 32;
#pragma unroll
    for (int i = 0; i < 4; ++i)
        t[ty + i*8][tx] = (_Float16)src[(size_t)(k0 + ty + i*8) * M + (m0 + tx)];
    __syncthreads();
#pragma unroll
    for (int i = 0; i < 4; ++i)
        dst[(size_t)(m0 + ty + i*8) * K + (k0 + tx)] = t[tx][ty + i*8];
}

// ---------------- batched f16 transpose: src[z][R][C] -> dst[z][C][R] --------
__global__ void k_tr_h(const _Float16* __restrict__ src, _Float16* __restrict__ dst,
                       int R, int C) {
    __shared__ _Float16 t[32][33];
    size_t zo = (size_t)blockIdx.z * R * C;
    int tx = threadIdx.x, ty = threadIdx.y;
    int c0 = blockIdx.x * 32, r0 = blockIdx.y * 32;
#pragma unroll
    for (int i = 0; i < 4; ++i)
        t[ty + i*8][tx] = src[zo + (size_t)(r0 + ty + i*8) * C + (c0 + tx)];
    __syncthreads();
#pragma unroll
    for (int i = 0; i < 4; ++i)
        dst[zo + (size_t)(c0 + ty + i*8) * R + (r0 + tx)] = t[tx][ty + i*8];
}

// ---------------- embedding + positional encoding ----------------------------
__global__ void k_embed(const int* __restrict__ src, const float* __restrict__ emb,
                        const float* __restrict__ pe, float* __restrict__ xf,
                        _Float16* __restrict__ xh) {
    int n = blockIdx.x;
    int d = blockIdx.y * 256 + threadIdx.x;
    int tok = src[n];
    float v = emb[(size_t)tok * DM + d] + pe[(size_t)(n % SEQ) * DM + d];
    xf[(size_t)n * DM + d] = v;
    xh[(size_t)n * DM + d] = (_Float16)v;
}

// ---------------- GEMM: C[N,M] = A[N,K](f16) @ Wt[M,K](f16)^T + bias ---------
// block: 128 thr = 4 waves; block tile 128(N-rows) x 64(M-cols); BK=32
// per-wave: 32 rows x 64 cols = 8 WMMA per K-step (2 A-frags x 4 B-frags)
// double-buffered LDS fed by async global->LDS DMA; one barrier per K-step.
__global__ __launch_bounds__(128)
void k_gemm(const _Float16* __restrict__ A, const _Float16* __restrict__ Wt,
            const float* __restrict__ bias, float* __restrict__ Cf,
            _Float16* __restrict__ Ch, int K, int M, int relu) {
    __shared__ _Float16 sA[2][128][40];   // padded rows: conflict-free b128 reads
    __shared__ _Float16 sB[2][64][40];
    int tid = threadIdx.x, w = tid >> 5, lane = tid & 31;
    int l15 = lane & 15, hi = lane >> 4;
    int n0 = blockIdx.y * 128, m0 = blockIdx.x * 64;

    v8f acc[2][4] = {};

    // staging: A tile 128x32 halves -> 1 row (4 x b128) per thread
    //          B tile  64x32 halves -> half row (2 x b128) per thread
    int arow = tid;
    int brow = tid >> 1, bch = (tid & 1) * 16;
    const _Float16* Ag = A  + (size_t)(n0 + arow) * K;
    const _Float16* Bg = Wt + (size_t)(m0 + brow) * K + bch;

    int fr = w * 32 + l15;    // A-frag rows: fr and fr+16
    int ab = hi ? 16 : 0;     // A-frag K-half select
    int bb = hi ? 16 : 0;     // B-frag K-half select

    auto issue_tile = [&](int buf, int ki) {
        const _Float16* As = Ag + ki * 32;
        const _Float16* Bs = Bg + ki * 32;
        async_ld_b128(&sA[buf][arow][0],  As);
        async_ld_b128(&sA[buf][arow][8],  As + 8);
        async_ld_b128(&sA[buf][arow][16], As + 16);
        async_ld_b128(&sA[buf][arow][24], As + 24);
        async_ld_b128(&sB[buf][brow][bch],     Bs);
        async_ld_b128(&sB[buf][brow][bch + 8], Bs + 8);
    };

    issue_tile(0, 0);

    int nk = K >> 5;
    for (int ki = 0; ki < nk; ++ki) {
        int cur = ki & 1;
        wait_async0();        // this thread's DMA for tile ki landed in LDS
        __syncthreads();      // everyone's tile-ki data visible; buf[cur^1] reads (iter ki-1) done
        if (ki + 1 < nk) issue_tile(cur ^ 1, ki + 1);   // DMA next tile during compute

        // load ALL fragments, then issue 8 WMMAs back-to-back
        v8h t0, t1;
        t0 = *(const v8h*)&sA[cur][fr][ab];
        t1 = *(const v8h*)&sA[cur][fr][ab + 8];
        v16h af0 = CAT16(t0, t1);
        t0 = *(const v8h*)&sA[cur][fr + 16][ab];
        t1 = *(const v8h*)&sA[cur][fr + 16][ab + 8];
        v16h af1 = CAT16(t0, t1);
        v16h bf[4];
#pragma unroll
        for (int j = 0; j < 4; ++j) {
            v8h y0 = *(const v8h*)&sB[cur][j * 16 + l15][bb];
            v8h y1 = *(const v8h*)&sB[cur][j * 16 + l15][bb + 8];
            bf[j] = CAT16(y0, y1);
        }
#pragma unroll
        for (int j = 0; j < 4; ++j) {
            acc[0][j] = WMMA(af0, bf[j], acc[0][j]);
            acc[1][j] = WMMA(af1, bf[j], acc[1][j]);
        }
    }

#pragma unroll
    for (int j = 0; j < 4; ++j) {
        int col = m0 + j * 16 + l15;
        float bv = bias[col];
#pragma unroll
        for (int i = 0; i < 2; ++i) {
#pragma unroll
            for (int r = 0; r < 8; ++r) {
                int row = n0 + w * 32 + i * 16 + r + hi * 8;
                float v = acc[i][j][r] + bv;
                if (relu) v = fmaxf(v, 0.0f);
                if (Cf) Cf[(size_t)row * M + col] = v;
                if (Ch) Ch[(size_t)row * M + col] = (_Float16)v;
            }
        }
    }
}

// ---------------- flash attention: 1 block = (b,h, 64 q-rows), 4 waves -------
__global__ __launch_bounds__(128, 2)
void k_attn(const _Float16* __restrict__ qh, const _Float16* __restrict__ kh,
            const _Float16* __restrict__ vt, _Float16* __restrict__ oh) {
    __shared__ _Float16 sP[4][16][40];
    int tid = threadIdx.x, w = tid >> 5, lane = tid & 31;
    int l15 = lane & 15, hi = lane >> 4;
    const int nq = SEQ / 64;
    int q64 = blockIdx.x % nq;
    int h   = (blockIdx.x / nq) % NH;
    int b   = blockIdx.x / (nq * NH);
    int hoff = h * DKH;
    int q0 = q64 * 64 + w * 16;
    int ab = hi ? 16 : 0;
    int bb = hi ? 16 : 0;
    const float scale = 0.125f;   // 1/sqrt(64)

    // Q A-fragments (d=0..31 and d=32..63), loaded once
    size_t qrow = (size_t)(b * SEQ + q0 + l15) * DM + hoff;
    v8h t0 = *(const v8h*)(qh + qrow + ab);
    v8h t1 = *(const v8h*)(qh + qrow + ab + 8);
    v16h qa0 = CAT16(t0, t1);
    t0 = *(const v8h*)(qh + qrow + 32 + ab);
    t1 = *(const v8h*)(qh + qrow + 40 + ab);
    v16h qa1 = CAT16(t0, t1);

    v8f o[4] = {v8f{}, v8f{}, v8f{}, v8f{}};
    float mrun[8], lsum[8];
#pragma unroll
    for (int r = 0; r < 8; ++r) { mrun[r] = -1e30f; lsum[r] = 0.0f; }

    size_t kbase = (size_t)(b * SEQ) * DM + hoff;
    size_t vbase = (size_t)((b * NH + h) * DKH) * SEQ;

    for (int kb = 0; kb < SEQ; kb += 32) {
        // scores tile 0: keys kb..kb+15
        size_t kr0 = kbase + (size_t)(kb + l15) * DM;
        v8h c0 = *(const v8h*)(kh + kr0 + bb);
        v8h c1 = *(const v8h*)(kh + kr0 + bb + 8);
        v8f s0 = {};
        s0 = WMMA(qa0, CAT16(c0, c1), s0);
        c0 = *(const v8h*)(kh + kr0 + 32 + bb);
        c1 = *(const v8h*)(kh + kr0 + 40 + bb);
        s0 = WMMA(qa1, CAT16(c0, c1), s0);
        // scores tile 1: keys kb+16..kb+31
        size_t kr1 = kr0 + (size_t)16 * DM;
        c0 = *(const v8h*)(kh + kr1 + bb);
        c1 = *(const v8h*)(kh + kr1 + bb + 8);
        v8f s1 = {};
        s1 = WMMA(qa0, CAT16(c0, c1), s1);
        c0 = *(const v8h*)(kh + kr1 + 32 + bb);
        c1 = *(const v8h*)(kh + kr1 + 40 + bb);
        s1 = WMMA(qa1, CAT16(c0, c1), s1);

        // hoist V B-frag global loads above the softmax VALU block
        v16h vb[4];
#pragma unroll
        for (int j = 0; j < 4; ++j) {
            size_t vr = vbase + (size_t)(j * 16 + l15) * SEQ + kb + bb;
            v8h v0 = *(const v8h*)(vt + vr);
            v8h v1 = *(const v8h*)(vt + vr + 8);
            vb[j] = CAT16(v0, v1);
        }

        // online softmax (rows live across 16-lane halves)
        float p0a[8], p1a[8];
#pragma unroll
        for (int r = 0; r < 8; ++r) {
            float a0 = s0[r] * scale, a1 = s1[r] * scale;
            float rmax = fmaxf(a0, a1);
            rmax = fmaxf(rmax, __shfl_xor(rmax, 8));
            rmax = fmaxf(rmax, __shfl_xor(rmax, 4));
            rmax = fmaxf(rmax, __shfl_xor(rmax, 2));
            rmax = fmaxf(rmax, __shfl_xor(rmax, 1));
            float mn = fmaxf(mrun[r], rmax);
            float corr = __expf(mrun[r] - mn);
            float p0 = __expf(a0 - mn), p1 = __expf(a1 - mn);
            float rs = p0 + p1;
            rs += __shfl_xor(rs, 8);
            rs += __shfl_xor(rs, 4);
            rs += __shfl_xor(rs, 2);
            rs += __shfl_xor(rs, 1);
            lsum[r] = lsum[r] * corr + rs;
            mrun[r] = mn;
            o[0][r] *= corr; o[1][r] *= corr; o[2][r] *= corr; o[3][r] *= corr;
            p0a[r] = p0; p1a[r] = p1;
        }
        // stage P (C-layout -> A-frag layout via LDS)
#pragma unroll
        for (int r = 0; r < 8; ++r) {
            sP[w][r + hi * 8][l15]      = (_Float16)p0a[r];
            sP[w][r + hi * 8][16 + l15] = (_Float16)p1a[r];
        }
        __syncthreads();
        v8h plo = *(const v8h*)&sP[w][l15][ab];
        v8h phi = *(const v8h*)&sP[w][l15][ab + 8];
        v16h pa = CAT16(plo, phi);
        // O += P @ V   (V pre-transposed: Vt[d][s])
#pragma unroll
        for (int j = 0; j < 4; ++j)
            o[j] = WMMA(pa, vb[j], o[j]);
        __syncthreads();
    }

#pragma unroll
    for (int r = 0; r < 8; ++r) {
        float inv = 1.0f / lsum[r];
        int srow = q0 + r + hi * 8;
        size_t base = (size_t)(b * SEQ + srow) * DM + hoff;
#pragma unroll
        for (int j = 0; j < 4; ++j)
            oh[base + j * 16 + l15] = (_Float16)(o[j][r] * inv);
    }
}

// ---------------- residual add + layernorm (block = 1 token, 256 thr) --------
__global__ __launch_bounds__(256)
void k_add_ln(const float* __restrict__ x, const float* __restrict__ res,
              const float* __restrict__ g, const float* __restrict__ bta,
              float* __restrict__ yf, _Float16* __restrict__ yh) {
    __shared__ float sh[2][8];
    int n = blockIdx.x, tid = threadIdx.x;
    size_t base = (size_t)n * DM;
    float v[3];
    float s = 0.0f, s2 = 0.0f;
#pragma unroll
    for (int i = 0; i < 3; ++i) {
        int d = tid + i * 256;
        float t = x[base + d];
        if (res) t += res[base + d];
        v[i] = t; s += t; s2 += t * t;
    }
#pragma unroll
    for (int msk = 16; msk >= 1; msk >>= 1) {
        s  += __shfl_xor(s,  msk);
        s2 += __shfl_xor(s2, msk);
    }
    int wid = tid >> 5, ln = tid & 31;
    if (ln == 0) { sh[0][wid] = s; sh[1][wid] = s2; }
    __syncthreads();
    float ts = 0.0f, ts2 = 0.0f;
#pragma unroll
    for (int i = 0; i < 8; ++i) { ts += sh[0][i]; ts2 += sh[1][i]; }
    float mu = ts / DM;
    float var = ts2 / DM - mu * mu;
    float rs = rsqrtf(var + 1e-5f);
#pragma unroll
    for (int i = 0; i < 3; ++i) {
        int d = tid + i * 256;
        float y = (v[i] - mu) * rs * g[d] + bta[d];
        yf[base + d] = y;
        if (yh) yh[base + d] = (_Float16)y;
    }
}

// ---------------- host orchestration -----------------------------------------
extern "C" void kernel_launch(void* const* d_in, const int* in_sizes, int n_in,
                              void* d_out, int out_size, void* d_ws, size_t ws_size,
                              hipStream_t stream) {
    const int*   src   = (const int*)  d_in[0];
    const float* emb   = (const float*)d_in[1];
    const float* pe    = (const float*)d_in[2];
    const float* Wq    = (const float*)d_in[3];
    const float* bq    = (const float*)d_in[4];
    const float* Wk    = (const float*)d_in[5];
    const float* bk    = (const float*)d_in[6];
    const float* Wv    = (const float*)d_in[7];
    const float* bv    = (const float*)d_in[8];
    const float* Wo    = (const float*)d_in[9];
    const float* bo    = (const float*)d_in[10];
    const float* ln1g  = (const float*)d_in[11];
    const float* ln1b  = (const float*)d_in[12];
    const float* W1    = (const float*)d_in[13];
    const float* b1    = (const float*)d_in[14];
    const float* W2    = (const float*)d_in[15];
    const float* b2    = (const float*)d_in[16];
    const float* ln2g  = (const float*)d_in[17];
    const float* ln2b  = (const float*)d_in[18];
    const float* fing  = (const float*)d_in[19];
    const float* finb  = (const float*)d_in[20];

    char* ws = (char*)d_ws;
    size_t off = 0;
    auto alloc = [&](size_t bytes) -> void* {
        void* p = ws + off;
        off += (bytes + 255) & ~(size_t)255;
        return p;
    };

    _Float16* wqt[NL]; _Float16* wkt[NL]; _Float16* wvt[NL]; _Float16* wot[NL];
    _Float16* w1t[NL]; _Float16* w2t[NL];
    for (int l = 0; l < NL; ++l) {
        wqt[l] = (_Float16*)alloc((size_t)DM * DM * 2);
        wkt[l] = (_Float16*)alloc((size_t)DM * DM * 2);
        wvt[l] = (_Float16*)alloc((size_t)DM * DM * 2);
        wot[l] = (_Float16*)alloc((size_t)DM * DM * 2);
        w1t[l] = (_Float16*)alloc((size_t)DM * DFF * 2);   // [DFF][DM]
        w2t[l] = (_Float16*)alloc((size_t)DFF * DM * 2);   // [DM][DFF]
    }
    float*    xf   = (float*)   alloc((size_t)NTOK * DM * 4);
    _Float16* xh   = (_Float16*)alloc((size_t)NTOK * DM * 2);
    _Float16* qh   = (_Float16*)alloc((size_t)NTOK * DM * 2);
    _Float16* khb  = (_Float16*)alloc((size_t)NTOK * DM * 2);
    _Float16* vhb  = (_Float16*)alloc((size_t)NTOK * DM * 2);
    _Float16* vtb  = (_Float16*)alloc((size_t)NTOK * DM * 2);
    _Float16* ohb  = (_Float16*)alloc((size_t)NTOK * DM * 2);
    _Float16* ffh  = (_Float16*)alloc((size_t)NTOK * DFF * 2);
    float*    scf  = (float*)   alloc((size_t)NTOK * DM * 4);

    dim3 trb(32, 8);
    for (int l = 0; l < NL; ++l) {
        k_tr_f32h<<<dim3(DM/32,  DM/32),  trb, 0, stream>>>(Wq + (size_t)l*DM*DM,  wqt[l], DM,  DM);
        k_tr_f32h<<<dim3(DM/32,  DM/32),  trb, 0, stream>>>(Wk + (size_t)l*DM*DM,  wkt[l], DM,  DM);
        k_tr_f32h<<<dim3(DM/32,  DM/32),  trb, 0, stream>>>(Wv + (size_t)l*DM*DM,  wvt[l], DM,  DM);
        k_tr_f32h<<<dim3(DM/32,  DM/32),  trb, 0, stream>>>(Wo + (size_t)l*DM*DM,  wot[l], DM,  DM);
        k_tr_f32h<<<dim3(DFF/32, DM/32),  trb, 0, stream>>>(W1 + (size_t)l*DM*DFF, w1t[l], DM,  DFF);
        k_tr_f32h<<<dim3(DM/32,  DFF/32), trb, 0, stream>>>(W2 + (size_t)l*DFF*DM, w2t[l], DFF, DM);
    }

    k_embed<<<dim3(NTOK, DM/256), 256, 0, stream>>>(src, emb, pe, xf, xh);

    dim3 gD(DM/64,  NTOK/128);   // GEMMs with M=768
    dim3 gF(DFF/64, NTOK/128);   // GEMM with M=3072
    int attnBlocks = NB * NH * (SEQ/64);

    for (int l = 0; l < NL; ++l) {
        k_gemm<<<gD, 128, 0, stream>>>(xh, wqt[l], bq + l*DM, nullptr, qh,  DM, DM, 0);
        k_gemm<<<gD, 128, 0, stream>>>(xh, wkt[l], bk + l*DM, nullptr, khb, DM, DM, 0);
        k_gemm<<<gD, 128, 0, stream>>>(xh, wvt[l], bv + l*DM, nullptr, vhb, DM, DM, 0);
        // V: [b][s][768] -> [b][768][s]  (== [b][h][dk][s])
        k_tr_h<<<dim3(DM/32, SEQ/32, NB), trb, 0, stream>>>(vhb, vtb, SEQ, DM);
        k_attn<<<attnBlocks, 128, 0, stream>>>(qh, khb, vtb, ohb);
        k_gemm<<<gD, 128, 0, stream>>>(ohb, wot[l], bo + l*DM, scf, nullptr, DM, DM, 0);
        k_add_ln<<<NTOK, 256, 0, stream>>>(xf, scf, ln1g + l*DM, ln1b + l*DM, xf, xh);
        k_gemm<<<gF, 128, 0, stream>>>(xh,  w1t[l], b1 + l*DFF, nullptr, ffh, DM,  DFF, 1);
        k_gemm<<<gD, 128, 0, stream>>>(ffh, w2t[l], b2 + l*DM,  scf, nullptr,  DFF, DM, 0);
        k_add_ln<<<NTOK, 256, 0, stream>>>(xf, scf, ln2g + l*DM, ln2b + l*DM, xf, xh);
    }

    k_add_ln<<<NTOK, 256, 0, stream>>>(xf, nullptr, fing, finb, (float*)d_out, nullptr);
}
